// Attention_335007449901
// MI455X (gfx1250) — matrix operand order repaired and checked by reading, not compiled, not run on hardware
//
#include <hip/hip_runtime.h>
#include <hip/hip_bf16.h>

typedef __bf16 bf16;
typedef __attribute__((ext_vector_type(8)))  __bf16 v8bf;
typedef __attribute__((ext_vector_type(16))) __bf16 v16bf;
typedef __attribute__((ext_vector_type(8)))  float  v8f;

#define NB    32
#define INC   128
#define HIDE_ 256
#define OUTC_ 128
#define WW    112
#define HW_   12544
#define WS_   7
#define NT    256        /* windows per image (16*16) */
#define CQK_  6272       /* 128*49 */
#define CV_   12544      /* 256*49 */
#define EPS_  1e-5f
#define SCALE_ 0.125f

/* workspace layout (bytes) */
#define SZ_QF   ((size_t)NB*NT*CQK_*2)       /* 102,760,448 */
#define SZ_VF   ((size_t)NB*CV_*NT*2)        /* 205,520,896 */
#define SZ_ATT  ((size_t)NB*NT*NT*2)
#define OFF_QF   ((size_t)0)
#define OFF_KF   (OFF_QF + SZ_QF)
#define OFF_VF   (OFF_KF + SZ_QF)
#define OFF_ATT  (OFF_VF + SZ_VF)
#define OFF_WQKV (OFF_ATT + SZ_ATT)
#define OFF_WO   (OFF_WQKV + (size_t)512*128*2)
#define OFF_SC512 (OFF_WO + (size_t)128*256*2)
#define OFF_BI512 (OFF_SC512 + (size_t)512*4)
#define OFF_SCO   (OFF_BI512 + (size_t)512*4)
#define OFF_BIO   (OFF_SCO + (size_t)128*4)
#define OFF_MID   ((size_t)0)                /* reuses qf+kf space (dead by then) */

__device__ __forceinline__ v8f wmma_bf16(v16bf a, v16bf b, v8f c) {
  return __builtin_amdgcn_wmma_f32_16x16x32_bf16(false, a, false, b, (short)0, c,
                                                 false, false);
}

/* A-matrix 16x32 bf16 fragment, source row-major [row][K], leading dim ld.
   Lane: row = lane&15; lanes 0-15 hold K = {0..7,16..23}, lanes 16-31 hold
   K = {8..15,24..31}. Two contiguous 16-byte loads + shuffle. */
__device__ __forceinline__ v16bf frag_a(const bf16* p, int ld, int lane) {
  const int row = lane & 15, kh = lane >> 4;
  const bf16* q = p + row * ld + kh * 8;
  v8bf lo = *(const v8bf*)q;
  v8bf hi = *(const v8bf*)(q + 16);
  return __builtin_shufflevector(lo, hi, 0,1,2,3,4,5,6,7,8,9,10,11,12,13,14,15);
}

/* B-matrix 32x16 bf16 fragment, B stored column-major as arr[n][k], leading dim ld.
   Lane: col = lane&15; lanes 0-15 hold K = {0..7,16..23}, lanes 16-31 hold
   K = {8..15,24..31}. Two contiguous 16-byte loads + shuffle. */
__device__ __forceinline__ v16bf frag_b(const bf16* p, int ld, int lane) {
  const int col = lane & 15, kh = lane >> 4;
  const bf16* q = p + col * ld + kh * 8;
  v8bf lo = *(const v8bf*)q;
  v8bf hi = *(const v8bf*)(q + 16);
  return __builtin_shufflevector(lo, hi, 0,1,2,3,4,5,6,7,8,9,10,11,12,13,14,15);
}

__device__ __forceinline__ float gelu_exact(float v) {
  return 0.5f * v * (1.0f + erff(v * 0.70710678118654752f));
}

/* ---------- prep: bf16 weights + folded BN scale/bias ---------- */
__global__ __launch_bounds__(256) void prep_kernel(
    const float* __restrict__ wq, const float* __restrict__ wk,
    const float* __restrict__ wv, const float* __restrict__ wo,
    const float* __restrict__ gq, const float* __restrict__ bq,
    const float* __restrict__ mq, const float* __restrict__ vq,
    const float* __restrict__ gk, const float* __restrict__ bk,
    const float* __restrict__ mk, const float* __restrict__ vk,
    const float* __restrict__ gv, const float* __restrict__ bv,
    const float* __restrict__ mv, const float* __restrict__ vvp,
    const float* __restrict__ bo, const float* __restrict__ go,
    const float* __restrict__ bbo, const float* __restrict__ mo,
    const float* __restrict__ vo,
    bf16* __restrict__ wqkv, bf16* __restrict__ wobf,
    float* __restrict__ sc512, float* __restrict__ bi512,
    float* __restrict__ sco, float* __restrict__ bio)
{
  const int t = blockIdx.x * 256 + threadIdx.x;
  if (t < 512 * 128) {
    const int o = t >> 7, c = t & 127;
    const float w = (o < 128) ? wq[o * 128 + c]
                  : (o < 256) ? wk[(o - 128) * 128 + c]
                              : wv[(o - 256) * 128 + c];
    wqkv[t] = (bf16)w;
  } else if (t < 512 * 128 + 128 * 256) {
    const int u = t - 512 * 128;
    wobf[u] = (bf16)wo[u];
  } else if (t < 512 * 128 + 128 * 256 + 512) {
    const int o = t - (512 * 128 + 128 * 256);
    float g, bb, mm, va;
    if (o < 128)      { g = gq[o]; bb = bq[o]; mm = mq[o]; va = vq[o]; }
    else if (o < 256) { const int c = o - 128; g = gk[c]; bb = bk[c]; mm = mk[c]; va = vk[c]; }
    else              { const int c = o - 256; g = gv[c]; bb = bv[c]; mm = mv[c]; va = vvp[c]; }
    const float s = g * rsqrtf(va + EPS_);
    sc512[o] = s;
    bi512[o] = bb - mm * s;
  } else if (t < 512 * 128 + 128 * 256 + 512 + 128) {
    const int o = t - (512 * 128 + 128 * 256 + 512);
    const float s = go[o] * rsqrtf(vo[o] + EPS_);
    sco[o] = s;
    bio[o] = s * (bo[o] - mo[o]) + bbo[o];
  }
}

/* ---------- qkv: [512x128] x [128 x 64 pixels] GEMM + BN + windowed store ----- */
__global__ __launch_bounds__(256) void qkv_kernel(
    const float* __restrict__ x, const bf16* __restrict__ wqkv,
    const float* __restrict__ sc512, const float* __restrict__ bi512,
    bf16* __restrict__ qf, bf16* __restrict__ kf, bf16* __restrict__ vf)
{
  __shared__ __align__(32) bf16 xs[64 * 128];   /* [pixel][channel], 16 KB */
  const int tid = threadIdx.x;
  const int gp0 = blockIdx.x * 64;              /* 12544 % 64 == 0: no batch straddle */
  const int b = gp0 / HW_;
  const int pbase = gp0 - b * HW_;

  { /* stage + convert x tile (coalesced over pixels) */
    const int pix = tid & 63;
    const int cb = tid >> 6;                    /* 0..3 */
    #pragma unroll
    for (int it = 0; it < 32; ++it) {
      const int c = cb * 32 + it;
      xs[pix * 128 + c] = (bf16)x[((size_t)b * INC + c) * HW_ + pbase + pix];
    }
  }
  __syncthreads();

  const int w = tid >> 5, lane = tid & 31;
  const int mrow = w * 64;                      /* 8 waves x 64 rows = 512 outputs */

  v8f acc[4][4];
  const v8f zero = {0.f,0.f,0.f,0.f,0.f,0.f,0.f,0.f};
  #pragma unroll
  for (int mi = 0; mi < 4; ++mi)
    #pragma unroll
    for (int ni = 0; ni < 4; ++ni) acc[mi][ni] = zero;

  #pragma unroll
  for (int kk = 0; kk < 128; kk += 32) {
    v16bf a[4], bfr[4];
    #pragma unroll
    for (int mi = 0; mi < 4; ++mi)
      a[mi] = frag_a(wqkv + (mrow + mi * 16) * 128 + kk, 128, lane);
    #pragma unroll
    for (int ni = 0; ni < 4; ++ni)
      bfr[ni] = frag_b(xs + ni * 16 * 128 + kk, 128, lane);
    #pragma unroll
    for (int mi = 0; mi < 4; ++mi)
      #pragma unroll
      for (int ni = 0; ni < 4; ++ni)
        acc[mi][ni] = wmma_bf16(a[mi], bfr[ni], acc[mi][ni]);
  }

  const int lh = lane >> 4, lc = lane & 15;
  #pragma unroll
  for (int mi = 0; mi < 4; ++mi) {
    const int obase = mrow + mi * 16 + lh * 8;
    #pragma unroll
    for (int ni = 0; ni < 4; ++ni) {
      const int p = pbase + ni * 16 + lc;
      const int y = p / WW, xx = p - y * WW;
      const int i = (y / WS_) * 16 + (xx / WS_);
      const int s = (y % WS_) * WS_ + (xx % WS_);
      #pragma unroll
      for (int r = 0; r < 8; ++r) {
        const int o = obase + r;
        const bf16 bvv = (bf16)(acc[mi][ni][r] * sc512[o] + bi512[o]);
        if (o < 128)
          qf[((size_t)b * NT + i) * CQK_ + o * 49 + s] = bvv;
        else if (o < 256)
          kf[((size_t)b * NT + i) * CQK_ + (o - 128) * 49 + s] = bvv;
        else
          vf[((size_t)b * CV_ + ((o - 256) * 49 + s)) * NT + i] = bvv;
      }
    }
  }
}

/* ---------- dots = qf^T kf (K=6272) * 0.125, softmax over 256 cols ---------- */
__global__ __launch_bounds__(256) void dots_softmax_kernel(
    const bf16* __restrict__ qf, const bf16* __restrict__ kf,
    bf16* __restrict__ attn)
{
  __shared__ float ds[32 * 256];                /* 32 KB */
  const int b = blockIdx.y;
  const int m0 = blockIdx.x * 32;
  const int tid = threadIdx.x;
  const int w = tid >> 5, lane = tid & 31;
  const int mloc = (w >> 2) * 16;               /* 0 or 16 */
  const int nfb = (w & 3) * 4;                  /* 4 n-frags per wave */

  v8f acc[4];
  const v8f zero = {0.f,0.f,0.f,0.f,0.f,0.f,0.f,0.f};
  #pragma unroll
  for (int ni = 0; ni < 4; ++ni) acc[ni] = zero;

  const bf16* qrow  = qf + ((size_t)b * NT + (m0 + mloc)) * CQK_;
  const bf16* kbase = kf + (size_t)b * NT * CQK_;

  for (int kk = 0; kk < CQK_; kk += 32) {
    /* stream prefetch ~32 K-steps ahead (global_prefetch_b8, speculative) */
    if (kk + 1024 < CQK_) {
      __builtin_prefetch(qrow + kk + 1024, 0, 1);
      __builtin_prefetch(kbase + (size_t)nfb * 16 * CQK_ + kk + 1024, 0, 1);
    }
    const v16bf a = frag_a(qrow + kk, CQK_, lane);
    #pragma unroll
    for (int ni = 0; ni < 4; ++ni) {
      const v16bf bfr = frag_b(kbase + (size_t)(nfb + ni) * 16 * CQK_ + kk, CQK_, lane);
      acc[ni] = wmma_bf16(a, bfr, acc[ni]);
    }
  }

  const int lh = lane >> 4, lc = lane & 15;
  #pragma unroll
  for (int ni = 0; ni < 4; ++ni)
    #pragma unroll
    for (int r = 0; r < 8; ++r)
      ds[(mloc + lh * 8 + r) * 256 + (nfb + ni) * 16 + lc] = acc[ni][r];
  __syncthreads();

  /* parallel softmax: 8 lanes per row (same wave), shfl_xor reductions */
  {
    const int r = tid >> 3;                     /* 0..31 */
    const int seg = tid & 7;                    /* 32 cols per lane */
    const float* row = ds + r * 256 + seg * 32;
    float mx = -3.0e38f;
    #pragma unroll
    for (int j = 0; j < 32; ++j) mx = fmaxf(mx, row[j] * SCALE_);
    #pragma unroll
    for (int d = 1; d < 8; d <<= 1) mx = fmaxf(mx, __shfl_xor(mx, d, 32));

    float e[32];
    float sum = 0.f;
    #pragma unroll
    for (int j = 0; j < 32; ++j) {
      e[j] = expf(row[j] * SCALE_ - mx);
      sum += e[j];
    }
    #pragma unroll
    for (int d = 1; d < 8; d <<= 1) sum += __shfl_xor(sum, d, 32);
    const float inv = 1.0f / sum;

    bf16* arow = attn + ((size_t)b * NT + m0 + r) * NT + seg * 32;
    #pragma unroll
    for (int j = 0; j < 32; ++j) arow[j] = (bf16)(e[j] * inv);
  }
}

/* ---------- out = attn @ vf^T, GELU, un-window into mid[b][pixel][c] ---------- */
__global__ __launch_bounds__(256) void av_kernel(
    const bf16* __restrict__ attn, const bf16* __restrict__ vf,
    bf16* __restrict__ mid)
{
  const int b = blockIdx.z;
  const int m0 = blockIdx.y * 64;
  const int n0 = blockIdx.x * 128;
  const int tid = threadIdx.x;
  const int w = tid >> 5, lane = tid & 31;
  const int mrow = m0 + (w >> 1) * 16;
  const int nb = n0 + (w & 1) * 64;

  v8f acc[4];
  const v8f zero = {0.f,0.f,0.f,0.f,0.f,0.f,0.f,0.f};
  #pragma unroll
  for (int ni = 0; ni < 4; ++ni) acc[ni] = zero;

  const bf16* arow  = attn + ((size_t)b * NT + mrow) * NT;
  const bf16* vbase = vf + (size_t)b * CV_ * NT;

  #pragma unroll
  for (int kk = 0; kk < 256; kk += 32) {
    const v16bf a = frag_a(arow + kk, NT, lane);
    #pragma unroll
    for (int ni = 0; ni < 4; ++ni) {
      const v16bf bfr = frag_b(vbase + (size_t)(nb + ni * 16) * NT + kk, NT, lane);
      acc[ni] = wmma_bf16(a, bfr, acc[ni]);
    }
  }

  const int lh = lane >> 4, lc = lane & 15;
  #pragma unroll
  for (int ni = 0; ni < 4; ++ni) {
    const int d = nb + ni * 16 + lc;
    const int c = d / 49, s = d - c * 49;
    const int sy = s / WS_, sx = s - sy * WS_;
    #pragma unroll
    for (int r = 0; r < 8; ++r) {
      const int i = mrow + lh * 8 + r;
      const int y  = (i >> 4) * WS_ + sy;
      const int xx = (i & 15) * WS_ + sx;
      mid[((size_t)b * HW_ + y * WW + xx) * HIDE_ + c] = (bf16)gelu_exact(acc[ni][r]);
    }
  }
}

/* ---------- final conv (128x256) + bias/BN + residual + GELU -> d_out -------- */
__global__ __launch_bounds__(256) void out_kernel(
    const bf16* __restrict__ mid, const bf16* __restrict__ wobf,
    const float* __restrict__ sco, const float* __restrict__ bio,
    const float* __restrict__ x, float* __restrict__ out)
{
  const int blk = blockIdx.x;                   /* 3136 = 32 batches * 98 tiles */
  const int b = blk / 98;
  const int pbase = (blk - b * 98) * 128;
  const int tid = threadIdx.x;
  const int w = tid >> 5, lane = tid & 31;
  const int mrow = w * 16;

  v8f acc[8];
  const v8f zero = {0.f,0.f,0.f,0.f,0.f,0.f,0.f,0.f};
  #pragma unroll
  for (int ni = 0; ni < 8; ++ni) acc[ni] = zero;

  const bf16* arow  = wobf + mrow * 256;
  const bf16* mbase = mid + ((size_t)b * HW_ + pbase) * HIDE_;

  #pragma unroll
  for (int kk = 0; kk < 256; kk += 32) {
    const v16bf a = frag_a(arow + kk, 256, lane);
    #pragma unroll
    for (int ni = 0; ni < 8; ++ni) {
      const v16bf bfr = frag_b(mbase + (size_t)(ni * 16) * HIDE_ + kk, HIDE_, lane);
      acc[ni] = wmma_bf16(a, bfr, acc[ni]);
    }
  }

  const int lh = lane >> 4, lc = lane & 15;
  #pragma unroll
  for (int ni = 0; ni < 8; ++ni) {
    const int pix = pbase + ni * 16 + lc;
    #pragma unroll
    for (int r = 0; r < 8; ++r) {
      const int o = mrow + lh * 8 + r;
      const size_t idx = ((size_t)b * OUTC_ + o) * HW_ + pix;
      float v = acc[ni][r] * sco[o] + bio[o];
      v += x[idx];
      out[idx] = gelu_exact(v);
    }
  }
}

extern "C" void kernel_launch(void* const* d_in, const int* in_sizes, int n_in,
                              void* d_out, int out_size, void* d_ws, size_t ws_size,
                              hipStream_t stream) {
  (void)in_sizes; (void)n_in; (void)out_size; (void)ws_size;
  const float* x   = (const float*)d_in[0];
  const float* wq  = (const float*)d_in[1];
  const float* gq  = (const float*)d_in[2];
  const float* bq  = (const float*)d_in[3];
  const float* mq  = (const float*)d_in[4];
  const float* vq  = (const float*)d_in[5];
  const float* wk  = (const float*)d_in[6];
  const float* gk  = (const float*)d_in[7];
  const float* bk  = (const float*)d_in[8];
  const float* mk  = (const float*)d_in[9];
  const float* vk  = (const float*)d_in[10];
  const float* wv  = (const float*)d_in[11];
  const float* gv  = (const float*)d_in[12];
  const float* bv  = (const float*)d_in[13];
  const float* mv  = (const float*)d_in[14];
  const float* vvp = (const float*)d_in[15];
  const float* wo  = (const float*)d_in[16];
  const float* bo  = (const float*)d_in[17];
  const float* go  = (const float*)d_in[18];
  const float* bbo = (const float*)d_in[19];
  const float* mo  = (const float*)d_in[20];
  const float* vo  = (const float*)d_in[21];

  char* ws = (char*)d_ws;
  bf16* qf    = (bf16*)(ws + OFF_QF);
  bf16* kfp   = (bf16*)(ws + OFF_KF);
  bf16* vfp   = (bf16*)(ws + OFF_VF);
  bf16* attn  = (bf16*)(ws + OFF_ATT);
  bf16* wqkv  = (bf16*)(ws + OFF_WQKV);
  bf16* wobf  = (bf16*)(ws + OFF_WO);
  float* sc512 = (float*)(ws + OFF_SC512);
  float* bi512 = (float*)(ws + OFF_BI512);
  float* sco   = (float*)(ws + OFF_SCO);
  float* bio   = (float*)(ws + OFF_BIO);
  bf16* mid   = (bf16*)(ws + OFF_MID);   /* recycles qf+kf region */

  prep_kernel<<<387, 256, 0, stream>>>(wq, wk, wv, wo, gq, bq, mq, vq,
                                       gk, bk, mk, vk, gv, bv, mv, vvp,
                                       bo, go, bbo, mo, vo,
                                       wqkv, wobf, sc512, bi512, sco, bio);
  qkv_kernel<<<(NB * HW_) / 64, 256, 0, stream>>>(x, wqkv, sc512, bi512,
                                                  qf, kfp, vfp);
  dots_softmax_kernel<<<dim3(8, NB), 256, 0, stream>>>(qf, kfp, attn);
  av_kernel<<<dim3(98, 4, NB), 256, 0, stream>>>(attn, vfp, mid);
  out_kernel<<<(NB * HW_) / 128, 256, 0, stream>>>(mid, wobf, sco, bio, x,
                                                   (float*)d_out);
}
